// DCA_91319594648302
// MI455X (gfx1250) — matrix operand order repaired and checked by reading, compiled, not run on hardware
//
#include <hip/hip_runtime.h>
#include <cstdint>

// ---------- types ----------
typedef __attribute__((ext_vector_type(16))) __bf16     v16bf;
typedef __attribute__((ext_vector_type(8)))  float      v8f;
typedef __attribute__((ext_vector_type(8)))  unsigned   v8u;

__device__ __forceinline__ unsigned short f2bf(float f) {
    unsigned u = __builtin_bit_cast(unsigned, f);
    return (unsigned short)((u + 0x7FFFu + ((u >> 16) & 1u)) >> 16);
}

// ---------- elementwise f32 -> bf16 convert (weights) ----------
__global__ void cvt_bf16_kernel(const float* __restrict__ in,
                                unsigned short* __restrict__ out, int n) {
    int i = blockIdx.x * 256 + threadIdx.x;
    if (i < n) out[i] = f2bf(in[i]);
}

__global__ void zero_f32_kernel(float* __restrict__ p, int n) {
    int i = blockIdx.x * 256 + threadIdx.x;
    if (i < n) p[i] = 0.f;
}

// ---------- depthwise 3x3x3 conv positional encoding: out = x + conv(x)+b ----------
__global__ void cpe_kernel(const float* __restrict__ x, const float* __restrict__ w,
                           const float* __restrict__ bias, float* __restrict__ out) {
    long long i = (long long)blockIdx.x * 256 + threadIdx.x; // B*8*32*32*384 total
    int c = (int)(i % 384);
    long long s = i / 384;
    int ww = (int)(s % 32); s /= 32;
    int hh = (int)(s % 32); s /= 32;
    int dd = (int)(s % 8);
    int b  = (int)(s / 8);
    float acc = bias[c];
    #pragma unroll
    for (int kd = 0; kd < 3; kd++) {
        int d2 = dd + kd - 1; if (d2 < 0 || d2 >= 8) continue;
        #pragma unroll
        for (int kh = 0; kh < 3; kh++) {
            int h2 = hh + kh - 1; if (h2 < 0 || h2 >= 32) continue;
            #pragma unroll
            for (int kw = 0; kw < 3; kw++) {
                int w2 = ww + kw - 1; if (w2 < 0 || w2 >= 32) continue;
                long long idx = ((((long long)b * 8 + d2) * 32 + h2) * 32 + w2) * 384 + c;
                acc += w[c * 27 + kd * 9 + kh * 3 + kw] * x[idx];
            }
        }
    }
    out[i] = x[i] + acc;
}

// ---------- LayerNorm(384) -> bf16 ----------
__global__ void ln_kernel(const float* __restrict__ in, const float* __restrict__ g,
                          const float* __restrict__ b, unsigned short* __restrict__ out) {
    __shared__ float red[128];
    int row = blockIdx.x, t = threadIdx.x;
    const float* p = in + (long long)row * 384;
    float v0 = p[t], v1 = p[t + 128], v2 = p[t + 256];
    red[t] = v0 + v1 + v2;
    __syncthreads();
    for (int o = 64; o > 0; o >>= 1) { if (t < o) red[t] += red[t + o]; __syncthreads(); }
    float mean = red[0] * (1.f / 384.f);
    __syncthreads();
    float d0 = v0 - mean, d1 = v1 - mean, d2 = v2 - mean;
    red[t] = d0 * d0 + d1 * d1 + d2 * d2;
    __syncthreads();
    for (int o = 64; o > 0; o >>= 1) { if (t < o) red[t] += red[t + o]; __syncthreads(); }
    float rstd = rsqrtf(red[0] * (1.f / 384.f) + 1e-5f);
    unsigned short* q = out + (long long)row * 384;
    q[t]       = f2bf(d0 * rstd * g[t]       + b[t]);
    q[t + 128] = f2bf(d1 * rstd * g[t + 128] + b[t + 128]);
    q[t + 256] = f2bf(d2 * rstd * g[t + 256] + b[t + 256]);
}

// ---------- tiled bf16 WMMA GEMM (double-buffered, 128x64 macro-tile, 2x2 tiles/wave) ----
// C[MxN] = A[MxK] * B[KxN] (+bias)(+gelu)(+res->f32)
// FLAGS: 1=bias, 2=gelu, 4=f32-out-with-residual (else bf16 out)
template <int FLAGS>
__global__ __launch_bounds__(256)
void gemm_bf16_kernel(const unsigned short* __restrict__ A,
                      const unsigned short* __restrict__ B,
                      const float* __restrict__ bias,
                      const float* __restrict__ res,
                      void* __restrict__ Out, int M, int N, int K) {
    // double-buffered stages: A 128 rows x 32 K (+pad), B 64 N-rows x 32 K (transposed)
    __shared__ __align__(16) unsigned short As[2 * 128 * 40];
    __shared__ __align__(16) unsigned short Bs[2 * 64 * 40];
    const int tid = threadIdx.x, lane = tid & 31, wave = tid >> 5;
    const int wm = wave >> 1, wn = wave & 1;            // 4x2 wave grid, 32x32 per wave
    const int bm = blockIdx.y * 128, bn = blockIdx.x * 64;
    const int ar = tid >> 2, ac = (tid & 3) * 8;        // A staging rows ar, ar+64
    const int bk = tid >> 3, bn8 = (tid & 7) * 8;       // B staging

    unsigned short tA0[8], tA1[8], tB[8];
    auto loadG = [&](int k0) {
        if (bm + ar < M) {
            *(uint4*)tA0 = *(const uint4*)(A + (long long)(bm + ar) * K + k0 + ac);
        } else {
            for (int j = 0; j < 8; j++) tA0[j] = 0;
        }
        if (bm + ar + 64 < M) {
            *(uint4*)tA1 = *(const uint4*)(A + (long long)(bm + ar + 64) * K + k0 + ac);
        } else {
            for (int j = 0; j < 8; j++) tA1[j] = 0;
        }
        if (bn + bn8 < N) {
            *(uint4*)tB = *(const uint4*)(B + (long long)(k0 + bk) * N + bn + bn8);
        } else {
            for (int j = 0; j < 8; j++) tB[j] = 0;
        }
    };
    auto storeL = [&](int buf) {
        unsigned short* pa = As + buf * (128 * 40);
        unsigned short* pb = Bs + buf * (64 * 40);
        #pragma unroll
        for (int j = 0; j < 8; j++) pa[ar * 40 + ac + j] = tA0[j];
        #pragma unroll
        for (int j = 0; j < 8; j++) pa[(ar + 64) * 40 + ac + j] = tA1[j];
        #pragma unroll
        for (int j = 0; j < 8; j++) pb[(bn8 + j) * 40 + bk] = tB[j];
    };

    v8f acc00 = {}, acc01 = {}, acc10 = {}, acc11 = {};
    loadG(0);
    storeL(0);
    __syncthreads();
    int cur = 0;
    for (int k0 = 0; k0 < K; k0 += 32) {
        const bool hasNext = (k0 + 32 < K);
        if (hasNext) loadG(k0 + 32);                    // overlap HBM/L2 with WMMAs below
        if (k0 + 64 < K) {                              // global_prefetch two stages ahead
            if (bm + ar < M) __builtin_prefetch(A + (long long)(bm + ar) * K + k0 + 64 + ac, 0, 0);
            if (bn + bn8 < N) __builtin_prefetch(B + (long long)(k0 + 64 + bk) * N + bn + bn8, 0, 0);
        }
        const unsigned* As32 = (const unsigned*)(As + cur * (128 * 40)); // row stride 20 dw
        const unsigned* Bs32 = (const unsigned*)(Bs + cur * (64 * 40));
        const int r = lane & 15, kh = (lane >> 4) * 4;   // dword units (K/2)
        v8u a0r, a1r;
        #pragma unroll
        for (int j = 0; j < 4; j++) {
            a0r[j]     = As32[(wm * 32 + r) * 20 + kh + j];
            a0r[j + 4] = As32[(wm * 32 + r) * 20 + 8 + kh + j];
            a1r[j]     = As32[(wm * 32 + 16 + r) * 20 + kh + j];
            a1r[j + 4] = As32[(wm * 32 + 16 + r) * 20 + 8 + kh + j];
        }
        const int cB = lane & 15, kb = (lane >> 4) * 4;  // dword units
        v8u b0r, b1r;
        #pragma unroll
        for (int v = 0; v < 8; v++) {
            b0r[v] = Bs32[(wn * 32 + cB) * 20 + kb + (v & 3) + (v >> 2) * 8];
            b1r[v] = Bs32[(wn * 32 + 16 + cB) * 20 + kb + (v & 3) + (v >> 2) * 8];
        }
        v16bf a0 = __builtin_bit_cast(v16bf, a0r);
        v16bf a1 = __builtin_bit_cast(v16bf, a1r);
        v16bf b0 = __builtin_bit_cast(v16bf, b0r);
        v16bf b1 = __builtin_bit_cast(v16bf, b1r);
        acc00 = __builtin_amdgcn_wmma_f32_16x16x32_bf16(false, a0, false, b0, (short)0, acc00, false, false);
        acc01 = __builtin_amdgcn_wmma_f32_16x16x32_bf16(false, a0, false, b1, (short)0, acc01, false, false);
        acc10 = __builtin_amdgcn_wmma_f32_16x16x32_bf16(false, a1, false, b0, (short)0, acc10, false, false);
        acc11 = __builtin_amdgcn_wmma_f32_16x16x32_bf16(false, a1, false, b1, (short)0, acc11, false, false);
        if (hasNext) storeL(cur ^ 1);   // safe: other buffer; its readers finished last barrier
        __syncthreads();
        cur ^= 1;
    }
    // epilogue (C/D layout: VGPR j -> M = j + 8*(lane>=16), N = lane&15)
    const int rb0 = bm + wm * 32 + 8 * (lane >> 4);
    const int rb1 = rb0 + 16;
    const int c0 = bn + wn * 32 + (lane & 15);
    const int c1 = c0 + 16;
    #pragma unroll
    for (int half = 0; half < 2; half++) {
        const int rbase = half ? rb1 : rb0;
        #pragma unroll
        for (int j = 0; j < 8; j++) {
            int row = rbase + j;
            if (row >= M) continue;
            float x0 = half ? acc10[j] : acc00[j];
            float x1 = half ? acc11[j] : acc01[j];
            if (FLAGS & 1) { if (c0 < N) x0 += bias[c0]; if (c1 < N) x1 += bias[c1]; }
            if (FLAGS & 2) {
                x0 = 0.5f * x0 * (1.f + erff(x0 * 0.70710678118654752f));
                x1 = 0.5f * x1 * (1.f + erff(x1 * 0.70710678118654752f));
            }
            long long o0 = (long long)row * N + c0, o1 = (long long)row * N + c1;
            if (FLAGS & 4) {
                float* Of = (float*)Out;
                if (c0 < N) Of[o0] = res[o0] + x0;
                if (c1 < N) Of[o1] = res[o1] + x1;
            } else {
                unsigned short* Ob = (unsigned short*)Out;
                if (c0 < N) Ob[o0] = f2bf(x0);
                if (c1 < N) Ob[o1] = f2bf(x1);
            }
        }
    }
}

// ---------- attention scores: att[h][c][d] = scale * sum_n k[n,c]*v[n,d], split-K ----------
__global__ __launch_bounds__(288)
void attn_score_kernel(const unsigned short* __restrict__ kv, float* __restrict__ att) {
    __shared__ __align__(16) unsigned short kT[48 * 68]; // [c][n], 64-n chunk (+pad)
    __shared__ __align__(16) unsigned short vT[48 * 68]; // [d][n]
    const int hl = blockIdx.x;                 // 0..31 : (b, g, hp)
    const int b = hl >> 3, g = (hl >> 2) & 1, hp = hl & 3;
    const int offk = g * 192 + hp * 48, offv = 384 + offk;
    const int nbeg = blockIdx.y * 1024;        // split-K over 8 blocks
    const int tid = threadIdx.x, lane = tid & 31, wave = tid >> 5; // 9 waves
    const int tm = wave / 3, tn = wave % 3;    // 3x3 tiles of 16x16 over 48x48
    v8f acc = {};
    for (int n0 = nbeg; n0 < nbeg + 1024; n0 += 64) {
        for (int idx = tid; idx < 64 * 96; idx += 288) {
            int n = idx / 96, cc = idx % 96;
            long long row = ((long long)b * 8192 + n0 + n) * 768;
            if (cc < 48) kT[cc * 68 + n] = kv[row + offk + cc];
            else         vT[(cc - 48) * 68 + n] = kv[row + offv + (cc - 48)];
        }
        __syncthreads();
        const unsigned* k32 = (const unsigned*)kT; // row stride 34 dwords
        const unsigned* v32 = (const unsigned*)vT;
        #pragma unroll
        for (int kk = 0; kk < 2; kk++) {
            const int r = lane & 15, kh = (lane >> 4) * 4;
            v8u araw;
            #pragma unroll
            for (int j = 0; j < 4; j++) {
                araw[j]     = k32[(tm * 16 + r) * 34 + kk * 16 + kh + j];
                araw[j + 4] = k32[(tm * 16 + r) * 34 + kk * 16 + 8 + kh + j];
            }
            const int cB = lane & 15, kb = (lane >> 4) * 4;
            v8u braw;
            #pragma unroll
            for (int v = 0; v < 8; v++)
                braw[v] = v32[(tn * 16 + cB) * 34 + kk * 16 + kb + (v & 3) + (v >> 2) * 8];
            acc = __builtin_amdgcn_wmma_f32_16x16x32_bf16(
                false, __builtin_bit_cast(v16bf, araw),
                false, __builtin_bit_cast(v16bf, braw), (short)0, acc, false, false);
        }
        __syncthreads();
    }
    const float scale = 0.144337567297406441f; // 48^-0.5
    const int rb = tm * 16 + 8 * (lane >> 4);
    const int col = tn * 16 + (lane & 15);
    #pragma unroll
    for (int j = 0; j < 8; j++)
        atomicAdd(&att[(long long)hl * 2304 + (rb + j) * 48 + col], acc[j] * scale);
}

// ---------- softmax over last dim (48), fp32, in place ----------
__global__ void softmax48_kernel(float* __restrict__ att) {
    int hl = blockIdx.x, t = threadIdx.x;
    if (t < 48) {
        float* row = att + (long long)hl * 2304 + (long long)t * 48;
        float m = row[0];
        for (int i = 1; i < 48; i++) m = fmaxf(m, row[i]);
        float e[48], s = 0.f;
        for (int i = 0; i < 48; i++) { e[i] = __expf(row[i] - m); s += e[i]; }
        float inv = 1.f / s;
        for (int i = 0; i < 48; i++) row[i] = e[i] * inv;
    }
}

// ---------- attention apply: y[n][c] = sum_d att[c][d] * q[n][d] (WMMA, K padded 48->64) ----------
__global__ __launch_bounds__(256)
void attn_apply_kernel(const unsigned short* __restrict__ q, const float* __restrict__ att,
                       unsigned short* __restrict__ y) {
    __shared__ __align__(16) unsigned short Aq[128 * 72]; // [n][d] padded K=64
    __shared__ __align__(16) unsigned short Bt[48 * 72];  // [c][d]
    const int hl = blockIdx.y;
    const int b = hl >> 3, g = (hl >> 2) & 1, hp = hl & 3;
    const int m0 = blockIdx.x * 128;
    const int tid = threadIdx.x, lane = tid & 31, wave = tid >> 5;
    for (int idx = tid; idx < 128 * 64; idx += 256) {
        int r = idx >> 6, d = idx & 63;
        unsigned short val = 0;
        if (d < 48) val = q[((long long)b * 8192 + m0 + r) * 96 + g * 48 + d];
        Aq[r * 72 + d] = val;
    }
    for (int idx = tid; idx < 48 * 64; idx += 256) {
        int c = idx >> 6, d = idx & 63;
        unsigned short val = 0;
        if (d < 48) val = f2bf(att[(long long)hl * 2304 + c * 48 + d]);
        Bt[c * 72 + d] = val;
    }
    __syncthreads();
    const unsigned* A32 = (const unsigned*)Aq; // row stride 36 dwords
    const unsigned* B32 = (const unsigned*)Bt;
    const long long ybase = (long long)b * 8192 * 384 + (long long)(hp * 2 + g) * 8192 * 48;
    #pragma unroll
    for (int tn = 0; tn < 3; tn++) {
        v8f acc = {};
        #pragma unroll
        for (int kk = 0; kk < 2; kk++) {
            const int r = lane & 15, kh = (lane >> 4) * 4;
            v8u araw;
            #pragma unroll
            for (int j = 0; j < 4; j++) {
                araw[j]     = A32[(wave * 16 + r) * 36 + kk * 16 + kh + j];
                araw[j + 4] = A32[(wave * 16 + r) * 36 + kk * 16 + 8 + kh + j];
            }
            const int cB = lane & 15, kb = (lane >> 4) * 4;
            v8u braw;
            #pragma unroll
            for (int v = 0; v < 8; v++)
                braw[v] = B32[(tn * 16 + cB) * 36 + kk * 16 + kb + (v & 3) + (v >> 2) * 8];
            acc = __builtin_amdgcn_wmma_f32_16x16x32_bf16(
                false, __builtin_bit_cast(v16bf, araw),
                false, __builtin_bit_cast(v16bf, braw), (short)0, acc, false, false);
        }
        const int rb = wave * 16 + 8 * (lane >> 4);
        const int c = tn * 16 + (lane & 15);
        #pragma unroll
        for (int j = 0; j < 8; j++) {
            int n = m0 + rb + j;
            y[ybase + (long long)n * 48 + c] = f2bf(acc[j]);
        }
    }
}

// ---------- host ----------
extern "C" void kernel_launch(void* const* d_in, const int* in_sizes, int n_in,
                              void* d_out, int out_size, void* d_ws, size_t ws_size,
                              hipStream_t stream) {
    const float* x       = (const float*)d_in[0];
    const float* cpe0_w  = (const float*)d_in[2];
    const float* cpe0_b  = (const float*)d_in[3];
    const float* cpe1_w  = (const float*)d_in[4];
    const float* cpe1_b  = (const float*)d_in[5];
    const float* norm1_g = (const float*)d_in[6];
    const float* norm1_b = (const float*)d_in[7];
    const float* q_w     = (const float*)d_in[8];
    const float* kv_w    = (const float*)d_in[9];
    const float* proj_w  = (const float*)d_in[10];
    const float* proj_b  = (const float*)d_in[11];
    const float* norm2_g = (const float*)d_in[12];
    const float* norm2_b = (const float*)d_in[13];
    const float* mlp_w1  = (const float*)d_in[14];
    const float* mlp_b1  = (const float*)d_in[15];
    const float* mlp_w2  = (const float*)d_in[16];
    const float* mlp_b2  = (const float*)d_in[17];

    const int M = 4 * 8192;   // 32768 rows
    const int C = 384;
    uintptr_t ws = (uintptr_t)d_ws;
    auto alloc = [&](size_t bytes) -> void* {
        void* p = (void*)ws;
        ws += (bytes + 255) & ~(size_t)255;
        return p;
    };
    float*          xf   = (float*)alloc((size_t)M * C * 4);
    float*          xf2  = (float*)alloc((size_t)M * C * 4);
    unsigned short* lnb  = (unsigned short*)alloc((size_t)M * C * 2);
    unsigned short* qb   = (unsigned short*)alloc((size_t)M * 96 * 2);
    unsigned short* kvb  = (unsigned short*)alloc((size_t)M * 768 * 2);
    float*          attb = (float*)alloc((size_t)32 * 2304 * 4);
    unsigned short* yb   = (unsigned short*)alloc((size_t)M * C * 2);
    unsigned short* hb   = (unsigned short*)alloc((size_t)M * 1536 * 2);
    unsigned short* qwb  = (unsigned short*)alloc((size_t)C * 96 * 2);
    unsigned short* kvwb = (unsigned short*)alloc((size_t)C * 768 * 2);
    unsigned short* pwb  = (unsigned short*)alloc((size_t)C * C * 2);
    unsigned short* w1b  = (unsigned short*)alloc((size_t)C * 1536 * 2);
    unsigned short* w2b  = (unsigned short*)alloc((size_t)1536 * C * 2);

    auto cvt = [&](const float* src, unsigned short* dst, int n) {
        cvt_bf16_kernel<<<(n + 255) / 256, 256, 0, stream>>>(src, dst, n);
    };
    cvt(q_w, qwb, C * 96);
    cvt(kv_w, kvwb, C * 768);
    cvt(proj_w, pwb, C * C);
    cvt(mlp_w1, w1b, C * 1536);
    cvt(mlp_w2, w2b, 1536 * C);
    zero_f32_kernel<<<(32 * 2304 + 255) / 256, 256, 0, stream>>>(attb, 32 * 2304);

    // CPE 0
    cpe_kernel<<<(size_t)M * C / 256, 256, 0, stream>>>(x, cpe0_w, cpe0_b, xf);
    // LN1 -> bf16
    ln_kernel<<<M, 128, 0, stream>>>(xf, norm1_g, norm1_b, lnb);
    // q / kv projections
    gemm_bf16_kernel<0><<<dim3(2, M / 128), 256, 0, stream>>>(lnb, qwb, nullptr, nullptr, qb, M, 96, C);
    gemm_bf16_kernel<0><<<dim3(12, M / 128), 256, 0, stream>>>(lnb, kvwb, nullptr, nullptr, kvb, M, 768, C);
    // channel attention
    attn_score_kernel<<<dim3(32, 8), 288, 0, stream>>>(kvb, attb);
    softmax48_kernel<<<32, 64, 0, stream>>>(attb);
    attn_apply_kernel<<<dim3(64, 32), 256, 0, stream>>>(qb, attb, yb);
    // proj + bias + residual(xf) -> xf2 (f32)
    gemm_bf16_kernel<5><<<dim3(6, M / 128), 256, 0, stream>>>(yb, pwb, proj_b, xf, xf2, M, C, C);
    // CPE 1 (xf2 -> xf, reuse)
    cpe_kernel<<<(size_t)M * C / 256, 256, 0, stream>>>(xf2, cpe1_w, cpe1_b, xf);
    // LN2 -> bf16
    ln_kernel<<<M, 128, 0, stream>>>(xf, norm2_g, norm2_b, lnb);
    // MLP: fc1 + bias + gelu -> bf16, fc2 + bias + residual(xf) -> d_out (f32)
    gemm_bf16_kernel<3><<<dim3(24, M / 128), 256, 0, stream>>>(lnb, w1b, mlp_b1, nullptr, hb, M, 1536, C);
    gemm_bf16_kernel<5><<<dim3(6, M / 128), 256, 0, stream>>>(hb, w2b, mlp_b2, xf, (float*)d_out, M, C, 1536);
}
